// SeqAttnMatchNoMask_15247133901013
// MI455X (gfx1250) — compile-verified, matrix operand order checked by reading, not run on hardware
//
#include <hip/hip_runtime.h>

// ---------------------------------------------------------------------------
// SeqAttnMatchNoMask for MI455X (gfx1250, wave32, WMMA f16->f32, TDM)
//
//   x_proj = relu(x @ W^T + b)   [32,1024,1024]   (f16 in ws)
//   y_proj = relu(y @ W^T + b)   [32,1024,1024]   (f16 in ws)
//   S      = x_proj @ y_proj^T   (per batch, 32-row stripes in 128KB LDS;
//                                 Q stripe DMA'd into LDS via tensor_load_to_lds)
//   alpha  = softmax(S)          (in-place in LDS, f32)
//   out    = alpha @ y           (via y^T f16 in ws, f32 accum)
//
// Workspace layout (192 MiB): [x_proj f16 64MiB][y_proj f16 64MiB][yT f16 64MiB]
// ---------------------------------------------------------------------------

typedef __attribute__((ext_vector_type(16))) _Float16 v16h;
typedef __attribute__((ext_vector_type(8)))  _Float16 v8h;
typedef __attribute__((ext_vector_type(8)))  float    v8f;
typedef __attribute__((ext_vector_type(4)))  unsigned int v4u;
typedef __attribute__((ext_vector_type(8)))  int      v8i;
typedef __attribute__((ext_vector_type(4)))  int      v4i;

#define HDIM 1024
#define LSEQ 1024
#define NBATCH 32

// Attention LDS layout (bytes):
//   [0,           131072)  S:     32 x 1024 f32 scores
//   [131072,      132608)  red:   384 f32 reduction scratch
//   [132608,      198144)  Qlds:  32 x 1024 f16 Q stripe (TDM destination)
#define SFLOATS      (32 * 1024)
#define REDFLOATS    384
#define QLDS_OFFSET  ((SFLOATS + REDFLOATS) * 4)
#define ATTN_LDS     (QLDS_OFFSET + 32 * 1024 * 2)

// ---- fragment loaders -----------------------------------------------------
// ISA 16-bit A/B layout (16x32, K-major per lane):
//   lanes 0-15 : row = lane,    K = kk+0..7   and kk+16..23
//   lanes 16-31: row = lane-16, K = kk+8..15  and kk+24..31
// Caller passes row already including (lane & 15).

__device__ __forceinline__ v16h frag_from_f32(const float* __restrict__ base,
                                              int row, int stride, int kk, int lane) {
  const float* p = base + (size_t)row * stride + kk + ((lane >> 4) << 3);
  v16h r;
#pragma unroll
  for (int i = 0; i < 8; ++i) {
    r[i]     = (_Float16)p[i];
    r[i + 8] = (_Float16)p[i + 16];
  }
  return r;
}

__device__ __forceinline__ v16h frag_from_f16(const _Float16* base,
                                              int row, int stride, int kk, int lane) {
  const _Float16* p = base + (size_t)row * stride + kk + ((lane >> 4) << 3);
  v8h a = *reinterpret_cast<const v8h*>(p);        // 16B contiguous
  v8h b = *reinterpret_cast<const v8h*>(p + 16);   // 16B contiguous
  return __builtin_shufflevector(a, b, 0, 1, 2, 3, 4, 5, 6, 7,
                                        8, 9, 10, 11, 12, 13, 14, 15);
}

// ---- kernel 1: proj = relu(in @ W^T + bias), f32 in -> f16 out -------------
// Block tile 128(M) x 128(N), 8 waves, each wave 64x32 (4 M-subtiles x 2 N-subtiles).
// All 6 fragments are loaded into arrays BEFORE the 8 WMMAs so the scheduler
// can batch the loads into one clause and issue WMMAs back-to-back (the
// WMMA->VALU WAR hazard then costs one drain per K-step, not one per WMMA).
__global__ void proj_kernel(const float* __restrict__ X,
                            const float* __restrict__ W,
                            const float* __restrict__ bias,
                            _Float16* __restrict__ out) {
  const int lane = threadIdx.x & 31;
  const int wave = threadIdx.x >> 5;
  const int wm = wave & 1, wn = wave >> 1;
  const int rowbase = blockIdx.y * 128 + wm * 64;
  const int colbase = blockIdx.x * 128 + wn * 32;
  const int l15 = lane & 15;

  const v8f vzero = {0.f, 0.f, 0.f, 0.f, 0.f, 0.f, 0.f, 0.f};
  v8f acc[4][2];
#pragma unroll
  for (int i = 0; i < 4; ++i)
#pragma unroll
    for (int j = 0; j < 2; ++j) acc[i][j] = vzero;

  for (int kk = 0; kk < HDIM; kk += 32) {
    if (kk + 64 < HDIM) {  // CDNA5 global_prefetch_b8 for the K step two ahead
      __builtin_prefetch(X + (size_t)(rowbase + l15) * HDIM + kk + 64, 0, 3);
      __builtin_prefetch(W + (size_t)(colbase + l15) * HDIM + kk + 64, 0, 3);
    }
    v16h bfrag[2];
    v16h afrag[4];
#pragma unroll
    for (int j = 0; j < 2; ++j)
      bfrag[j] = frag_from_f32(W, colbase + j * 16 + l15, HDIM, kk, lane);
#pragma unroll
    for (int i = 0; i < 4; ++i)
      afrag[i] = frag_from_f32(X, rowbase + i * 16 + l15, HDIM, kk, lane);
#pragma unroll
    for (int i = 0; i < 4; ++i)
#pragma unroll
      for (int j = 0; j < 2; ++j)
        acc[i][j] = __builtin_amdgcn_wmma_f32_16x16x32_f16(
            false, afrag[i], false, bfrag[j], (short)0, acc[i][j], false, false);
  }

  // epilogue: bias + relu, f16 store. C/D layout: lane n=l15, vgpr v -> m=v+8*(lane>>4)
#pragma unroll
  for (int j = 0; j < 2; ++j) {
    const int n = colbase + j * 16 + l15;
    const float bv = bias[n];
#pragma unroll
    for (int i = 0; i < 4; ++i) {
#pragma unroll
      for (int v = 0; v < 8; ++v) {
        const int m = rowbase + i * 16 + v + ((lane >> 4) << 3);
        float val = acc[i][j][v] + bv;
        val = val > 0.f ? val : 0.f;
        out[(size_t)m * HDIM + n] = (_Float16)val;
      }
    }
  }
}

// ---- kernel 2: yT[b,h,j] = (f16) y[b,j,h], LDS-staged 64x64 tiles ----------
__global__ void transpose_y_kernel(const float* __restrict__ Y,
                                   _Float16* __restrict__ YT) {
  __shared__ _Float16 tile[64 * 72];  // pad stride 72 to break bank conflicts
  const int b  = blockIdx.z;
  const int jb = blockIdx.x * 64;
  const int hb = blockIdx.y * 64;
  const int t  = threadIdx.x;
  const int r  = t >> 2;          // 0..63
  const int c0 = (t & 3) * 16;    // 0,16,32,48

  const float* src = Y + (size_t)b * LSEQ * HDIM + (size_t)(jb + r) * HDIM + hb + c0;
#pragma unroll
  for (int i = 0; i < 16; ++i)
    tile[(c0 + i) * 72 + r] = (_Float16)src[i];   // [h_local][j_local]
  __syncthreads();
  _Float16* dst = YT + (size_t)b * HDIM * LSEQ + (size_t)(hb + r) * LSEQ + jb + c0;
#pragma unroll
  for (int i = 0; i < 16; ++i)
    dst[i] = tile[r * 72 + c0 + i];
}

// ---- TDM helper: DMA one contiguous 64KB Q stripe into LDS -----------------
// 1-D tile: 32 rows x 1024 f16 of x_proj are contiguous (32768 u16 elements).
// D# per CDNA5 ISA: group0 = {count|flags, lds_addr,
// global_addr lo, global_addr hi | type=2<<30}; group1 packs data_size=2B,
// tensor_dim0 = tile_dim0 = 32768, tensor_dim1 = 1.
__device__ __forceinline__ void tdm_load_q(const _Float16* qsrc) {
  const unsigned long long ga = (unsigned long long)(size_t)qsrc;
  v4u g0;
  g0[0] = 1u;                                   // count=1, user descriptor
  g0[1] = (unsigned)QLDS_OFFSET;                // lds_addr (bytes)
  g0[2] = (unsigned)(ga & 0xffffffffu);         // global_addr[31:0]
  g0[3] = (unsigned)((ga >> 32) & 0x01ffffffu)  // global_addr[56:32]
          | 0x80000000u;                        // type=2 ("image") at [127:126]
  v8i g1;
  g1[0] = 0x00010000;                           // wg_mask=0, data_size=1 (2 bytes)
  g1[1] = (int)0x80000000u;                     // tensor_dim0[15:0]=0x8000 at [63:48]
  g1[2] = 1 << 16;                              // tensor_dim0 hi=0, tensor_dim1=1
  g1[3] = (int)0x80000000u;                     // tile_dim0=0x8000 at [127:112]
  g1[4] = 0;                                    // tile_dim1=0 (unused), tile_dim2=0
  g1[5] = 32768;                                // tensor_dim0_stride lo
  g1[6] = 0;
  g1[7] = 0;
  v4i gz = {0, 0, 0, 0};
#if defined(__clang_major__) && (__clang_major__ >= 23)
  v8i gz8 = {0, 0, 0, 0, 0, 0, 0, 0};
  __builtin_amdgcn_tensor_load_to_lds(g0, g1, gz, gz, gz8, 0);
#else
  __builtin_amdgcn_tensor_load_to_lds(g0, g1, gz, gz, 0);
#endif
}

// ---- kernel 3: attention stripe ------------------------------------------
// One block = (batch, 32 rows of x). 8 waves x 256 threads.
// LDS: S[32][1024] f32 (128KB) + scratch (1.5KB) + Q[32][1024] f16 (64KB).
__global__ void attn_kernel(const _Float16* __restrict__ XP,
                            const _Float16* __restrict__ YP,
                            const _Float16* __restrict__ YT,
                            float* __restrict__ Out) {
  extern __shared__ float S[];            // 32*1024 scores
  float* red    = S + SFLOATS;            // 256 partials
  float* rowmax = red + 256;              // 32
  float* rowsum = rowmax + 32;            // 32 (stores 1/sum)
  const _Float16* Qlds = (const _Float16*)(S + SFLOATS + REDFLOATS);

  const int b    = blockIdx.x >> 5;
  const int rb   = (blockIdx.x & 31) * 32;
  const int lane = threadIdx.x & 31;
  const int wave = threadIdx.x >> 5;
  const int l15  = lane & 15;
  const int nc   = wave * 128;            // this wave's 128-wide N stripe

  // DMA the contiguous Q stripe (32 rows of x_proj) into LDS via the TDM.
  if (wave == 0) {
    tdm_load_q(XP + ((size_t)b * LSEQ + rb) * HDIM);
    __builtin_amdgcn_s_wait_tensorcnt(0);
  }
  __syncthreads();

  const _Float16* K = YP + (size_t)b * LSEQ * HDIM;
  const v8f vzero = {0.f, 0.f, 0.f, 0.f, 0.f, 0.f, 0.f, 0.f};

  // Phase 1: S[0:32, nc:nc+128] = Q @ K^T  (A from LDS, B from global)
  for (int ns = 0; ns < 8; ++ns) {
    const int col = nc + ns * 16;         // j base
    v8f acc[2];
#pragma unroll
    for (int i = 0; i < 2; ++i) acc[i] = vzero;
    for (int kk = 0; kk < HDIM; kk += 32) {
      v16h bfrag = frag_from_f16(K, col + l15, HDIM, kk, lane);
      v16h afrag[2];
#pragma unroll
      for (int i = 0; i < 2; ++i)
        afrag[i] = frag_from_f16(Qlds, i * 16 + l15, HDIM, kk, lane);  // ds_load_b128
#pragma unroll
      for (int i = 0; i < 2; ++i)
        acc[i] = __builtin_amdgcn_wmma_f32_16x16x32_f16(
            false, afrag[i], false, bfrag, (short)0, acc[i], false, false);
    }
#pragma unroll
    for (int i = 0; i < 2; ++i)
#pragma unroll
      for (int v = 0; v < 8; ++v) {
        const int m = i * 16 + v + ((lane >> 4) << 3);
        S[m * 1024 + col + l15] = acc[i][v];
      }
  }
  __syncthreads();

  // Softmax over each of the 32 rows: 8 threads per row, 128 cols each.
  {
    const int t = threadIdx.x;
    const int r = t >> 3;
    const int q = t & 7;
    float* row = S + r * 1024 + q * 128;
    float m = -3.4e38f;
    for (int i = 0; i < 128; ++i) m = fmaxf(m, row[i]);
    red[t] = m;
    __syncthreads();
    if (q == 0) {
      float mm = red[t];
#pragma unroll
      for (int i = 1; i < 8; ++i) mm = fmaxf(mm, red[t + i]);
      rowmax[r] = mm;
    }
    __syncthreads();
    const float mx = rowmax[r];
    float s = 0.f;
    for (int i = 0; i < 128; ++i) {
      const float e = __expf(row[i] - mx);
      row[i] = e;
      s += e;
    }
    red[t] = s;
    __syncthreads();
    if (q == 0) {
      float ss = red[t];
#pragma unroll
      for (int i = 1; i < 8; ++i) ss += red[t + i];
      rowsum[r] = 1.0f / ss;
    }
    __syncthreads();
    const float inv = rowsum[r];
    for (int i = 0; i < 128; ++i) row[i] *= inv;
  }
  __syncthreads();

  // Phase 2: Out[0:32, nc:nc+128] = alpha(LDS) @ y  (via yT, K-dim = LSEQ)
  const _Float16* V = YT + (size_t)b * HDIM * LSEQ;   // [h][j]
  float* O = Out + ((size_t)b * LSEQ + rb) * HDIM;
  for (int ns = 0; ns < 8; ++ns) {
    const int col = nc + ns * 16;         // h base
    v8f acc[2];
#pragma unroll
    for (int i = 0; i < 2; ++i) acc[i] = vzero;
    for (int kk = 0; kk < LSEQ; kk += 32) {
      v16h bfrag = frag_from_f16(V, col + l15, LSEQ, kk, lane);
      v16h afrag[2];
#pragma unroll
      for (int i = 0; i < 2; ++i) {
        const float* p = S + (i * 16 + l15) * 1024 + kk + ((lane >> 4) << 3);
#pragma unroll
        for (int e = 0; e < 8; ++e) {
          afrag[i][e]     = (_Float16)p[e];
          afrag[i][e + 8] = (_Float16)p[e + 16];
        }
      }
#pragma unroll
      for (int i = 0; i < 2; ++i)
        acc[i] = __builtin_amdgcn_wmma_f32_16x16x32_f16(
            false, afrag[i], false, bfrag, (short)0, acc[i], false, false);
    }
#pragma unroll
    for (int i = 0; i < 2; ++i)
#pragma unroll
      for (int v = 0; v < 8; ++v) {
        const int m = i * 16 + v + ((lane >> 4) << 3);
        O[(size_t)m * HDIM + col + l15] = acc[i][v];
      }
  }
}

// ---------------------------------------------------------------------------
extern "C" void kernel_launch(void* const* d_in, const int* in_sizes, int n_in,
                              void* d_out, int out_size, void* d_ws, size_t ws_size,
                              hipStream_t stream) {
  (void)in_sizes; (void)n_in; (void)out_size; (void)ws_size;
  const float* x    = (const float*)d_in[0];   // [32,1024,1024]
  const float* y    = (const float*)d_in[1];   // [32,1024,1024]
  const float* W    = (const float*)d_in[2];   // [1024,1024]
  const float* bias = (const float*)d_in[3];   // [1024]
  float* out = (float*)d_out;

  char* ws = (char*)d_ws;
  const size_t projBytes = (size_t)NBATCH * LSEQ * HDIM * sizeof(_Float16); // 64 MiB
  _Float16* xp = (_Float16*)(ws);
  _Float16* yp = (_Float16*)(ws + projBytes);
  _Float16* yt = (_Float16*)(ws + 2 * projBytes);

  // projections: M = 32*1024 rows, N = 1024
  dim3 gProj(HDIM / 128, (NBATCH * LSEQ) / 128);  // (8, 256)
  proj_kernel<<<gProj, 256, 0, stream>>>(x, W, bias, xp);
  proj_kernel<<<gProj, 256, 0, stream>>>(y, W, bias, yp);

  // y transpose to f16
  dim3 gT(LSEQ / 64, HDIM / 64, NBATCH);
  transpose_y_kernel<<<gT, 256, 0, stream>>>(y, yt);

  // attention stripes: 32 batches x 32 row-blocks, ~194KB dynamic LDS
  attn_kernel<<<NBATCH * (LSEQ / 32), 256, ATTN_LDS, stream>>>(xp, yp, yt, out);
}
